// PatchGCN_60224031425186
// MI455X (gfx1250) — matrix operand order repaired and checked by reading, not compiled, not run on hardware
//
#include <hip/hip_runtime.h>
#include <hip/hip_bf16.h>
#include <math.h>

// ---------------------------------------------------------------------------
// PatchGCN forward for MI455X (gfx1250, wave32, WMMA).
// Dense GEMMs: bf16 operands, f32 accumulate via v_wmma_f32_16x16x32_bf16.
// - 4 waves/block cover a 64x16 C tile; Kx16 B panel staged once into LDS
//   (bf16, pre-swizzled into the WMMA B lane layout -> one 32B ds read/k-step)
// - A can be read directly as bf16 (2x global_load_b128, no converts); GEMM
//   epilogues dual-write bf16 copies of C so chained GEMMs stay in bf16.
// ---------------------------------------------------------------------------

typedef __attribute__((ext_vector_type(16))) __bf16 v16bf;
typedef __attribute__((ext_vector_type(8)))  float  v8f;

#define ACT_NONE 0
#define ACT_RELU 1
#define ACT_TANH 2
#define ACT_SIG  3

// ---------------------------------------------------------------------------
// WMMA GEMM: C[M,N] = act(A[M,K] @ B + bias).
// ABF: A (and B when TRANSB) is bf16 instead of f32.
// TRANSB: B is [N,K] (only used for sim = e @ e^T, bf16).
// CBF: additionally write bf16 copy of C to Cbf (same ldc).
// M multiple of 64, N multiple of 16, K multiple of 32, K <= 512.
// Grid: (M/64)*(N/16) blocks of 128 threads.
// ---------------------------------------------------------------------------
template<int ABF, int TRANSB, int ACT, int CBF>
__global__ __launch_bounds__(128)
void wmma_gemm_kernel(const void* __restrict__ Av, int lda,
                      const void* __restrict__ Bv,
                      const float* __restrict__ bias,
                      float* __restrict__ C, int ldc,
                      __bf16* __restrict__ Cbf,
                      int M, int N, int K)
{
    __shared__ v16bf sBv[512];               // K(<=512) x 16 bf16 panel, 16 KB
    __bf16* sB = (__bf16*)sBv;

    int ntiles = N >> 4;
    int bm = blockIdx.x / ntiles;            // 64-row group
    int bn = blockIdx.x - bm * ntiles;
    int n0 = bn << 4;
    int tid = threadIdx.x;

    // Stage B panel into the WMMA B operand layout:
    //   element (k, n): lane L = n + 16*((k>>3)&1), elem e = (k&7) + 8*((k>>4)&1), block k>>5
    for (int idx = tid; idx < (K << 4); idx += 128) {
        int n = idx & 15;
        int k = idx >> 4;
        __bf16 bv;
        if (TRANSB) {
            if (ABF) bv = ((const __bf16*)Bv)[(size_t)(n0 + n) * K + k];
            else     bv = (__bf16)((const float*)Bv)[(size_t)(n0 + n) * K + k];
        } else {
            bv = (__bf16)((const float*)Bv)[(size_t)k * N + (n0 + n)];
        }
        int L = n + (((k >> 3) & 1) << 4);
        sB[((k >> 5) << 9) + (L << 4) + (k & 7) + (((k >> 4) & 1) << 3)] = bv;
    }
    __syncthreads();

    int wave = tid >> 5;
    int lane = tid & 31;
    int half = lane >> 4;                    // 0: lanes 0-15, 1: lanes 16-31
    int lq   = lane & 15;
    int m0 = (bm << 6) + (wave << 4);
    if (m0 >= M) return;

    v8f acc = {0.f,0.f,0.f,0.f,0.f,0.f,0.f,0.f};
    const float*  arowf = (const float*)Av  + (size_t)(m0 + lq) * lda;
    const __bf16* arowh = (const __bf16*)Av + (size_t)(m0 + lq) * lda;

    for (int kb = 0; kb < (K >> 5); ++kb) {
        int k0 = kb << 5;
        v16bf a;
        if (ABF) {
            if (k0 + 32 < K) __builtin_prefetch(arowh + k0 + 32, 0, 1);
            // lane row = lq; elems 0-7 -> K = k0 + half*8 + e (16B chunk),
            //                elems 8-15 -> K = k0 + 16 + half*8 + e (16B chunk)
#pragma unroll
            for (int e = 0; e < 8; ++e) a[e]     = arowh[k0 + (half << 3) + e];
#pragma unroll
            for (int e = 0; e < 8; ++e) a[e + 8] = arowh[k0 + 16 + (half << 3) + e];
        } else {
            if (k0 + 32 < K) __builtin_prefetch(arowf + k0 + 32, 0, 1);
#pragma unroll
            for (int e = 0; e < 16; ++e) {
                int ka = k0 + e + ((e & 8) ? 8 : 0) + (half << 3);
                a[e] = (__bf16)arowf[ka];
            }
        }
        v16bf b = sBv[(kb << 5) + lane];     // contiguous 32B LDS read
        acc = __builtin_amdgcn_wmma_f32_16x16x32_bf16(
                  false, a, false, b, (short)0, acc, false, false);
    }

    float bb = bias ? bias[n0 + lq] : 0.f;
#pragma unroll
    for (int r = 0; r < 8; ++r) {
        float v = acc[r] + bb;
        if      (ACT == ACT_RELU) v = fmaxf(v, 0.f);
        else if (ACT == ACT_TANH) v = tanhf(v);
        else if (ACT == ACT_SIG)  v = 1.f / (1.f + __expf(-v));
        int row = m0 + r + (half << 3);
        C[(size_t)row * ldc + (n0 + lq)] = v;
        if (CBF) Cbf[(size_t)row * ldc + (n0 + lq)] = (__bf16)v;
    }
}

// ---------------------------------------------------------------------------
// Utility / elementwise kernels
// ---------------------------------------------------------------------------
__global__ void fill_kernel(float* p, float v, size_t n)
{
    size_t i = (size_t)blockIdx.x * blockDim.x + threadIdx.x;
    if (i < n) p[i] = v;
}

__global__ void f32_to_bf16_kernel(const float* __restrict__ in,
                                   __bf16* __restrict__ out, size_t n)
{
    size_t i = (size_t)blockIdx.x * blockDim.x + threadIdx.x;
    if (i < n) out[i] = (__bf16)in[i];
}

// GENConv edge pass 1: per-(dst,channel) max of msg*t (all values > 0 ->
// uint atomicMax on the raw bits is an exact positive-float max).
__global__ void edge_max_kernel(const float* __restrict__ x, int ld,
                                const int* __restrict__ src,
                                const int* __restrict__ dst,
                                const float* __restrict__ t,
                                unsigned* __restrict__ mx, int E)
{
    int i = blockIdx.x * blockDim.x + threadIdx.x;
    if (i >= E * 128) return;
    int c = i & 127, e = i >> 7;
    float msg = fmaxf(x[(size_t)src[e] * ld + c], 0.f) + 1e-7f;
    atomicMax(&mx[(size_t)dst[e] * 128 + c], __float_as_uint(msg * t[0]));
}

// GENConv edge pass 2: accumulate softmax numerator/denominator.
__global__ void edge_sum_kernel(const float* __restrict__ x, int ld,
                                const int* __restrict__ src,
                                const int* __restrict__ dst,
                                const float* __restrict__ t,
                                const float* __restrict__ mx,
                                float* __restrict__ den,
                                float* __restrict__ num, int E)
{
    int i = blockIdx.x * blockDim.x + threadIdx.x;
    if (i >= E * 128) return;
    int c = i & 127, e = i >> 7;
    float msg = fmaxf(x[(size_t)src[e] * ld + c], 0.f) + 1e-7f;
    size_t o = (size_t)dst[e] * 128 + c;
    float ex = __expf(msg * t[0] - mx[o]);
    atomicAdd(&den[o], ex);
    atomicAdd(&num[o], ex * msg);
}

// GENConv combine: h = softmax-aggregated message + x (residual input to MLP)
__global__ void edge_combine_kernel(const float* __restrict__ x, int ld,
                                    const float* __restrict__ den,
                                    const float* __restrict__ num,
                                    float* __restrict__ out, int Nn)
{
    int i = blockIdx.x * blockDim.x + threadIdx.x;
    if (i >= Nn * 128) return;
    int c = i & 127, n = i >> 7;
    float aggr = num[i] / (den[i] + 1e-16f);
    out[i] = aggr + x[(size_t)n * ld + c];
}

// LayerNorm (+ReLU, optional residual): out = resid + relu(LN(in)*g + b)
__global__ void ln_relu_kernel(const float* __restrict__ in, int ldin, int W,
                               const float* __restrict__ g,
                               const float* __restrict__ b,
                               const float* __restrict__ resid, int ldres,
                               float* __restrict__ out, int ldout, int rows)
{
    int row = blockIdx.x * (blockDim.x >> 5) + (threadIdx.x >> 5);
    if (row >= rows) return;
    int lane = threadIdx.x & 31;
    const float* p = in + (size_t)row * ldin;
    float s = 0.f, ss = 0.f;
    for (int c = lane; c < W; c += 32) { float v = p[c]; s += v; ss += v * v; }
    for (int off = 16; off; off >>= 1) {
        s  += __shfl_xor(s,  off, 32);
        ss += __shfl_xor(ss, off, 32);
    }
    float mean = s / W;
    float inv  = rsqrtf(ss / W - mean * mean + 1e-5f);
    float*       q = out + (size_t)row * ldout;
    const float* r = resid ? resid + (size_t)row * ldres : nullptr;
    for (int c = lane; c < W; c += 32) {
        float v = fmaxf((p[c] - mean) * inv * g[c] + b[c], 0.f);
        q[c] = (r ? r[c] : 0.f) + v;
    }
}

// Gated-attention score: A[p] = sum_d tanh_a[p,d]*sig_g[p,d]*cw[d] + cb
__global__ void attn_score_kernel(const float* __restrict__ a,
                                  const float* __restrict__ g,
                                  const float* __restrict__ cw,
                                  const float* __restrict__ cb,
                                  float* __restrict__ scores, int rows)
{
    int p = blockIdx.x * (blockDim.x >> 5) + (threadIdx.x >> 5);
    if (p >= rows) return;
    int lane = threadIdx.x & 31;
    float s = 0.f;
    for (int d = lane; d < 512; d += 32)
        s += a[(size_t)p * 512 + d] * g[(size_t)p * 512 + d] * cw[d];
    for (int off = 16; off; off >>= 1) s += __shfl_xor(s, off, 32);
    if (lane == 0) scores[p] = s + cb[0];
}

// Per-bag softmax over 500 patches + attention-weighted pooling.
__global__ __launch_bounds__(512)
void bag_pool_kernel(const float* __restrict__ hp,
                     const float* __restrict__ scores,
                     float* __restrict__ pooled)
{
    __shared__ float sw[512];
    __shared__ float red[512];
    int b = blockIdx.x, tid = threadIdx.x;
    float lv = (tid < 500) ? scores[b * 500 + tid] : -3.0e38f;
    red[tid] = lv; __syncthreads();
    for (int off = 256; off; off >>= 1) {
        if (tid < off) red[tid] = fmaxf(red[tid], red[tid + off]);
        __syncthreads();
    }
    float mx = red[0]; __syncthreads();
    float ex = (tid < 500) ? __expf(lv - mx) : 0.f;
    sw[tid] = ex; red[tid] = ex; __syncthreads();
    for (int off = 256; off; off >>= 1) {
        if (tid < off) red[tid] += red[tid + off];
        __syncthreads();
    }
    float s = red[0]; __syncthreads();
    if (tid < 500) sw[tid] = sw[tid] / s;
    __syncthreads();
    const float* base = hp + (size_t)b * 500 * 512;
    float acc = 0.f;
    for (int n = 0; n < 500; ++n) acc += sw[n] * base[(size_t)n * 512 + tid];
    pooled[b * 512 + tid] = acc;
}

__global__ void out_mlp_kernel(const float* __restrict__ bag,
                               const float* __restrict__ W,
                               const float* __restrict__ bias,
                               float* __restrict__ out)
{
    int i = threadIdx.x >> 1, j = threadIdx.x & 1;   // 128 threads
    float s = bias[j];
    for (int d = 0; d < 512; ++d) s += bag[i * 512 + d] * W[d * 2 + j];
    out[i * 2 + j] = s;
}

__global__ void build_xc_kernel(const float* __restrict__ bag,
                                const float* __restrict__ reh,
                                float* __restrict__ xc)
{
    int i = blockIdx.x * blockDim.x + threadIdx.x;   // 1088*512
    if (i >= 1088 * 512) return;
    int row = i >> 9, c = i & 511;
    xc[i] = (row < 64) ? bag[row * 512 + c] : reh[(row - 64) * 512 + c];
}

// Wave-per-row top-12 over 1088 cols (34 vals/lane) with self-edge excluded,
// followed by softmax over the 12 kept values.
__global__ void topk_kernel(const float* __restrict__ sim,
                            float* __restrict__ attr,
                            int* __restrict__ kidx, int Mrows)
{
    int row = blockIdx.x * (blockDim.x >> 5) + (threadIdx.x >> 5);
    if (row >= Mrows) return;
    int lane = threadIdx.x & 31;
    float v[34];
    const float* p = sim + (size_t)row * 1088;
#pragma unroll
    for (int q = 0; q < 34; ++q) {
        int col = lane + (q << 5);
        float xv = p[col];
        v[q] = (col == row) ? -3.0e38f : xv;
    }
    float tv[12]; int ti[12];
    for (int t = 0; t < 12; ++t) {
        float bv = -3.0e38f; int bq = 0;
#pragma unroll
        for (int q = 0; q < 34; ++q) if (v[q] > bv) { bv = v[q]; bq = q; }
        float mv = bv; int ml = lane;
        for (int off = 16; off; off >>= 1) {
            float ov = __shfl_xor(mv, off, 32);
            int   ol = __shfl_xor(ml, off, 32);
            if (ov > mv || (ov == mv && ol < ml)) { mv = ov; ml = ol; }
        }
        int wq  = __shfl(bq, ml, 32);
        int col = ml + (wq << 5);
        if (lane == ml) v[wq] = -3.0e38f;
        tv[t] = mv; ti[t] = col;
    }
    float mmax = tv[0];
    for (int t = 1; t < 12; ++t) mmax = fmaxf(mmax, tv[t]);
    float s = 0.f, exv[12];
    for (int t = 0; t < 12; ++t) { exv[t] = __expf(tv[t] - mmax); s += exv[t]; }
    if (lane == 0)
        for (int t = 0; t < 12; ++t) {
            attr[row * 12 + t] = exv[t] / s;
            kidx[row * 12 + t] = ti[t];
        }
}

__global__ void deg_edge_kernel(const float* __restrict__ attr,
                                float* __restrict__ deg, int n)
{
    int i = blockIdx.x * blockDim.x + threadIdx.x;
    if (i < n) atomicAdd(&deg[i / 12], attr[i]);
}

__global__ void dinv_kernel(const float* __restrict__ deg,
                            float* __restrict__ dinv, int M)
{
    int i = blockIdx.x * blockDim.x + threadIdx.x;
    if (i < M) dinv[i] = rsqrtf(deg[i] + 1e-12f);
}

// Sym-normalized weighted GCN propagation. Every kNN edge of row i has dst==i,
// so prop is a pure per-row gather (12 neighbors + self loop).
// Rows >= srcLimit of h are treated as zero (zero-padded bag features).
__global__ __launch_bounds__(512)
void prop_kernel(const float* __restrict__ h, int C, int srcLimit,
                 const int* __restrict__ kidx,
                 const float* __restrict__ attr,
                 const float* __restrict__ dinv,
                 float* __restrict__ out)
{
    int i = blockIdx.x, c = threadIdx.x;
    if (c >= C) return;
    float di  = dinv[i];
    float acc = (i < srcLimit) ? h[(size_t)i * C + c] * di * di : 0.f;
    for (int t = 0; t < 12; ++t) {
        int s = kidx[i * 12 + t];
        if (s < srcLimit) {
            float w = attr[i * 12 + t] * dinv[s] * di;
            acc += w * h[(size_t)s * C + c];
        }
    }
    out[(size_t)i * C + c] = acc;
}

__global__ void out_graph_kernel(const float* __restrict__ p2,
                                 const float* __restrict__ W,
                                 const float* __restrict__ bias,
                                 float* __restrict__ out)
{
    int i = threadIdx.x >> 1, j = threadIdx.x & 1;   // 128 threads
    float s = bias[j];
    for (int d = 0; d < 256; ++d) s += p2[i * 256 + d] * W[d * 2 + j];
    out[128 + i * 2 + j] = s;
}

// ---------------------------------------------------------------------------
// Host-side orchestration
// ---------------------------------------------------------------------------
static inline void fill(hipStream_t st, float* p, float v, size_t n)
{
    fill_kernel<<<(unsigned)((n + 255) / 256), 256, 0, st>>>(p, v, n);
}

static inline dim3 gemm_grid(int M, int N) { return dim3((M / 64) * (N / 16)); }

extern "C" void kernel_launch(void* const* d_in, const int* in_sizes, int n_in,
                              void* d_out, int out_size, void* d_ws, size_t ws_size,
                              hipStream_t stream)
{
    (void)in_sizes; (void)n_in; (void)out_size; (void)ws_size;
    const int Nn = 32000, E = 256000, Bb = 64, Mg = 1088;

    // Inputs: insertion-order flattening of setup_inputs() pytree.
    const float* x   = (const float*)d_in[0];
    const int*   src = (const int*)d_in[1];
    const int*   dst = src + E;
    struct GenP { const float *t,*w1,*b1,*lng,*lnb,*w2,*b2; } gen[3];
    for (int l = 0; l < 3; ++l) {
        int base = 5 + l * 7;
        gen[l].t   = (const float*)d_in[base + 0];
        gen[l].w1  = (const float*)d_in[base + 1];
        gen[l].b1  = (const float*)d_in[base + 2];
        gen[l].lng = (const float*)d_in[base + 3];
        gen[l].lnb = (const float*)d_in[base + 4];
        gen[l].w2  = (const float*)d_in[base + 5];
        gen[l].b2  = (const float*)d_in[base + 6];
    }
    const float* fc_w  = (const float*)d_in[3];
    const float* fc_b  = (const float*)d_in[4];
    const float* dln_g[2] = { (const float*)d_in[26], (const float*)d_in[28] };
    const float* dln_b[2] = { (const float*)d_in[27], (const float*)d_in[29] };
    const float* phi_w = (const float*)d_in[30];
    const float* phi_b = (const float*)d_in[31];
    const float* aw    = (const float*)d_in[32];
    const float* ab    = (const float*)d_in[33];
    const float* bw    = (const float*)d_in[34];
    const float* bb    = (const float*)d_in[35];
    const float* cw    = (const float*)d_in[36];
    const float* cb    = (const float*)d_in[37];
    const float* rho_w = (const float*)d_in[38];
    const float* rho_b = (const float*)d_in[39];
    const float* cls_w = (const float*)d_in[40];
    const float* cls_b = (const float*)d_in[41];
    const float* dsl_w = (const float*)d_in[42];
    const float* dsl_b = (const float*)d_in[43];
    const float* g1w   = (const float*)d_in[44];
    const float* g1b   = (const float*)d_in[45];
    const float* g2w   = (const float*)d_in[46];
    const float* g2b   = (const float*)d_in[47];
    const float* reh   = (const float*)d_in[48];
    float* out = (float*)d_out;

    // Workspace layout (float-slot offsets).
    // Region [16384000, 45056000) is GENConv scratch, later reused by hp and
    // xcat_bf (both only live after GENConv is done).
    float* W = (float*)d_ws;
    float*  xcat    = W;                        // 32000*512 f32
    float*  mx      = W + 16384000;             // 32000*128 f32
    float*  den     = W + 20480000;             // 32000*128 f32
    float*  num     = W + 24576000;             // 32000*128 f32
    float*  hbuf    = W + 28672000;             // 32000*128 f32
    float*  hmid    = W + 32768000;             // 32000*256 f32
    float*  cbuf    = W + 40960000;             // 32000*128 f32
    float*  hp      = W + 16384000;             // 32000*512 f32 (reuse)
    __bf16* xcat_bf = (__bf16*)(W + 32768000);  // 32000*512 bf16 (reuse)
    float*  abuf    = W + 45056000;             // 32000*512 f32
    float*  gbuf    = W + 61440000;             // 32000*512 f32
    float*  scores  = W + 77824000;             // 32000
    float*  pooled  = W + 77856000;             // 64*512
    float*  bag     = W + 77888768;             // 64*512
    float*  xc      = W + 77921536;             // 1088*512
    float*  ebuf    = W + 78478592;             // 1088*512
    float*  sim     = W + 79035648;             // 1088*1088
    float*  attr    = W + 80219392;             // 1088*12
    int*    kidx    = (int*)(W + 80232448);     // 1088*12
    float*  deg     = W + 80245504;             // 1088
    float*  dinvb   = W + 80246592;             // 1088
    float*  prop1   = W + 80247680;             // 1088*512
    float*  h1g     = W + 80804736;             // 1088*256
    float*  prop2   = W + 81083264;             // 1088*256
    __bf16* hp_bf   = (__bf16*)(W + 81361792);  // 32000*512 bf16
    __bf16* ebuf_bf = (__bf16*)(W + 89553792);  // 1088*512 bf16

    const int EC = E * 128;
    const int NC = Nn * 128;

    // 1) fc: h0 = relu(x @ fc_w + fc_b) -> xcat[:, 0:128]
    wmma_gemm_kernel<0,0,ACT_RELU,0><<<gemm_grid(Nn,128), 128, 0, stream>>>(
        x, 512, fc_w, fc_b, xcat, 512, nullptr, Nn, 128, 512);

    // 2) three GENConv layers
    for (int l = 0; l < 3; ++l) {
        const float* xin = xcat + l * 128;      // ld 512
        fill(stream, mx, 0.f, NC);
        fill(stream, den, 0.f, NC);
        fill(stream, num, 0.f, NC);
        edge_max_kernel<<<(EC + 255) / 256, 256, 0, stream>>>(
            xin, 512, src, dst, gen[l].t, (unsigned*)mx, E);
        edge_sum_kernel<<<(EC + 255) / 256, 256, 0, stream>>>(
            xin, 512, src, dst, gen[l].t, mx, den, num, E);
        edge_combine_kernel<<<(NC + 255) / 256, 256, 0, stream>>>(
            xin, 512, den, num, hbuf, Nn);
        wmma_gemm_kernel<0,0,ACT_NONE,0><<<gemm_grid(Nn,256), 128, 0, stream>>>(
            hbuf, 128, gen[l].w1, gen[l].b1, hmid, 256, nullptr, Nn, 256, 128);
        ln_relu_kernel<<<(Nn + 7) / 8, 256, 0, stream>>>(
            hmid, 256, 256, gen[l].lng, gen[l].lnb, nullptr, 0, hmid, 256, Nn);
        if (l == 0) {
            wmma_gemm_kernel<0,0,ACT_NONE,0><<<gemm_grid(Nn,128), 128, 0, stream>>>(
                hmid, 256, gen[l].w2, gen[l].b2, xcat + 128, 512, nullptr, Nn, 128, 256);
        } else {
            wmma_gemm_kernel<0,0,ACT_NONE,0><<<gemm_grid(Nn,128), 128, 0, stream>>>(
                hmid, 256, gen[l].w2, gen[l].b2, cbuf, 128, nullptr, Nn, 128, 256);
            ln_relu_kernel<<<(Nn + 7) / 8, 256, 0, stream>>>(
                cbuf, 128, 128, dln_g[l - 1], dln_b[l - 1],
                xcat + l * 128, 512, xcat + (l + 1) * 128, 512, Nn);
        }
    }

    // 3) MIL attention tower (bf16 activations between chained GEMMs)
    f32_to_bf16_kernel<<<(Nn * 512 + 255) / 256, 256, 0, stream>>>(
        xcat, xcat_bf, (size_t)Nn * 512);
    wmma_gemm_kernel<1,0,ACT_RELU,1><<<gemm_grid(Nn,512), 128, 0, stream>>>(
        xcat_bf, 512, phi_w, phi_b, hp, 512, hp_bf, Nn, 512, 512);
    wmma_gemm_kernel<1,0,ACT_TANH,0><<<gemm_grid(Nn,512), 128, 0, stream>>>(
        hp_bf, 512, aw, ab, abuf, 512, nullptr, Nn, 512, 512);
    wmma_gemm_kernel<1,0,ACT_SIG,0><<<gemm_grid(Nn,512), 128, 0, stream>>>(
        hp_bf, 512, bw, bb, gbuf, 512, nullptr, Nn, 512, 512);
    attn_score_kernel<<<(Nn + 7) / 8, 256, 0, stream>>>(abuf, gbuf, cw, cb, scores, Nn);
    bag_pool_kernel<<<Bb, 512, 0, stream>>>(hp, scores, pooled);
    wmma_gemm_kernel<0,0,ACT_RELU,0><<<gemm_grid(Bb,512), 128, 0, stream>>>(
        pooled, 512, rho_w, rho_b, bag, 512, nullptr, Bb, 512, 512);
    out_mlp_kernel<<<1, 128, 0, stream>>>(bag, cls_w, cls_b, out);

    // 4) DSL: dynamic kNN slide graph + 2-layer weighted GCN
    build_xc_kernel<<<(Mg * 512 + 255) / 256, 256, 0, stream>>>(bag, reh, xc);
    wmma_gemm_kernel<0,0,ACT_NONE,1><<<gemm_grid(Mg,512), 128, 0, stream>>>(
        xc, 512, dsl_w, dsl_b, ebuf, 512, ebuf_bf, Mg, 512, 512);
    wmma_gemm_kernel<1,1,ACT_NONE,0><<<gemm_grid(Mg,Mg), 128, 0, stream>>>(
        ebuf_bf, 512, ebuf_bf, nullptr, sim, Mg, nullptr, Mg, Mg, 512);
    topk_kernel<<<(Mg + 7) / 8, 256, 0, stream>>>(sim, attr, kidx, Mg);
    fill(stream, deg, 1.f, Mg);                 // self-loop weight
    deg_edge_kernel<<<(Mg * 12 + 255) / 256, 256, 0, stream>>>(attr, deg, Mg * 12);
    dinv_kernel<<<(Mg + 255) / 256, 256, 0, stream>>>(deg, dinvb, Mg);
    prop_kernel<<<Mg, 512, 0, stream>>>(bag, 512, Bb, kidx, attr, dinvb, prop1);
    wmma_gemm_kernel<0,0,ACT_RELU,0><<<gemm_grid(Mg,256), 128, 0, stream>>>(
        prop1, 512, g1w, g1b, h1g, 256, nullptr, Mg, 256, 512);
    prop_kernel<<<Mg, 256, 0, stream>>>(h1g, 256, Mg, kidx, attr, dinvb, prop2);
    out_graph_kernel<<<1, 128, 0, stream>>>(prop2, g2w, g2b, out);
}
